// OT_Loss_54614804136622
// MI455X (gfx1250) — matrix operand order repaired and checked by reading, not compiled, not run on hardware
//
#include <hip/hip_runtime.h>
#include <math.h>

// ---------------------------------------------------------------------------
// Problem: N=M=512 points, dim K=16384 (batch treated as the point cloud).
// Sinkhorn p=2, blur=0.01, reach=0.3 (rho=0.09), scaling=0.5, diameter=1.
// ---------------------------------------------------------------------------
#define NPTS 512
#define KDIM 16384
#define RHO_C 0.09f

typedef __attribute__((ext_vector_type(8)))  __bf16 bf16x8;
typedef __attribute__((ext_vector_type(16))) __bf16 bf16x16;
typedef __attribute__((ext_vector_type(8)))  float  f32x8;
typedef __attribute__((ext_vector_type(4)))  int    i32x4;

using gv4i = i32x4 __attribute__((address_space(1)));   // global int4
using lv4i = i32x4 __attribute__((address_space(3)));   // LDS int4

// ---- CDNA5 async global->LDS path (ASYNCcnt), with safe fallback -----------
#if defined(__has_builtin)
#if __has_builtin(__builtin_amdgcn_global_load_async_to_lds_b128)
#define ASYNC_LDS 1
#endif
#endif
#ifndef ASYNC_LDS
#define ASYNC_LDS 0
#endif

__device__ __forceinline__ void copy16_to_lds(const __bf16* g, __bf16* l) {
#if ASYNC_LDS
    __builtin_amdgcn_global_load_async_to_lds_b128(
        (gv4i*)(uintptr_t)g,   // AS1: 64-bit address, inttoptr is exact
        (lv4i*)l,              // generic -> AS3 addrspacecast
        0, 0);
#else
    *(float4*)l = *(const float4*)g;
#endif
}

__device__ __forceinline__ void wait_stage() {
#if ASYNC_LDS
#if __has_builtin(__builtin_amdgcn_s_wait_asynccnt)
    __builtin_amdgcn_s_wait_asynccnt(0);
#else
    asm volatile("s_wait_asynccnt 0" ::: "memory");
#endif
#endif
}

// ---------------------------------------------------------------------------
// Kernel 1: fp32 -> bf16 conversion of x and y, fused with row squared-norms.
// grid.x = 1024 (512 rows of x, then 512 rows of y), block = 256.
// ---------------------------------------------------------------------------
__global__ void convert_norm_kernel(const float* __restrict__ x,
                                    const float* __restrict__ y,
                                    __bf16* __restrict__ xb,
                                    __bf16* __restrict__ yb,
                                    float* __restrict__ nx,
                                    float* __restrict__ ny) {
    const int row = blockIdx.x & (NPTS - 1);
    const int isY = blockIdx.x >> 9;
    const float* __restrict__ src = (isY ? y : x) + (size_t)row * KDIM;
    __bf16* __restrict__ dst = (isY ? yb : xb) + (size_t)row * KDIM;

    float acc = 0.0f;
    for (int j = threadIdx.x; j < KDIM; j += 256) {
        float v = src[j];
        acc += v * v;
        dst[j] = (__bf16)v;
    }
    __shared__ float red[256];
    red[threadIdx.x] = acc;
    __syncthreads();
    for (int s = 128; s > 0; s >>= 1) {
        if (threadIdx.x < s) red[threadIdx.x] += red[threadIdx.x + s];
        __syncthreads();
    }
    if (threadIdx.x == 0) (isY ? ny : nx)[row] = red[0];
}

// ---------------------------------------------------------------------------
// Kernel 2: LDS-blocked cost-matrix GEMMs via v_wmma_f32_16x16x32_bf16.
//   C[i][j] = 0.5*(|a_i|^2 + |b_j|^2) - a_i . b_j
// Workgroup (256 thr = 8 waves) computes a 64x64 tile of C; K staged through
// LDS in chunks of 64, double-buffered via async global->LDS loads.
// Each wave owns two 16x16 WMMA tiles (shared A fragment).
// grid = (64, 3): grid.y selects {C_xy, C_xx, C_yy}.
// ---------------------------------------------------------------------------
#define BM 64
#define BN 64
#define KC 64
#define LSTR 72   // 64 + 8 pad -> 144B row stride (9x16B): rotates LDS banks

__global__ void __launch_bounds__(256)
gemm_cost_kernel(const __bf16* __restrict__ xb, const __bf16* __restrict__ yb,
                 const float* __restrict__ nx, const float* __restrict__ ny,
                 float* __restrict__ Cxy, float* __restrict__ Cxx,
                 float* __restrict__ Cyy) {
    __shared__ __align__(16) __bf16 Abuf[2][BM][LSTR];
    __shared__ __align__(16) __bf16 Bbuf[2][BN][LSTR];

    const __bf16* A; const __bf16* B; const float* na; const float* nb; float* Cout;
    switch (blockIdx.y) {
        case 0:  A = xb; B = yb; na = nx; nb = ny; Cout = Cxy; break;
        case 1:  A = xb; B = xb; na = nx; nb = nx; Cout = Cxx; break;
        default: A = yb; B = yb; na = ny; nb = ny; Cout = Cyy; break;
    }

    const int bm0 = (blockIdx.x >> 3) * BM;
    const int bn0 = (blockIdx.x & 7) * BN;
    const int t    = threadIdx.x;
    const int lane = t & 31;
    const int wv   = t >> 5;        // 0..7
    const int half = lane >> 4;     // 0/1
    const int r    = lane & 15;
    const int tm   = wv & 3;        // A-tile row 0..3
    const int tn2  = wv >> 2;       // 0/1 -> B cols {tn2*32, tn2*32+16}

    // Panel staging: 64 rows x 64 bf16 per matrix = 512 16B-chunks;
    // thread t copies chunks {t, t+256} of A and of B.
    auto stage = [&](int buf, int k0) {
#pragma unroll
        for (int p = 0; p < 2; ++p) {
            const int c   = t + p * 256;
            const int row = c >> 3;
            const int ko  = (c & 7) * 8;
            copy16_to_lds(A + (size_t)(bm0 + row) * KDIM + k0 + ko, &Abuf[buf][row][ko]);
            copy16_to_lds(B + (size_t)(bn0 + row) * KDIM + k0 + ko, &Bbuf[buf][row][ko]);
        }
    };

    f32x8 acc0 = {};
    f32x8 acc1 = {};
    const int NS = KDIM / KC;   // 256 stages

    stage(0, 0);
    int buf = 0;
    for (int s = 0; s < NS; ++s) {
        wait_stage();           // this wave's panel bytes are in LDS
        __syncthreads();        // everyone's are; prior reads of buf^1 done
        if (s + 1 < NS) stage(buf ^ 1, (s + 1) * KC);
#pragma unroll
        for (int k = 0; k < KC; k += 32) {
            // A fragment: contiguous k order (k = 16*half + s), same as B
            const __bf16* ar = &Abuf[buf][tm * 16 + r][k + 16 * half];
            bf16x8 a0 = *(const bf16x8*)ar;
            bf16x8 a1 = *(const bf16x8*)(ar + 8);
            bf16x16 a = __builtin_shufflevector(a0, a1,
                        0,1,2,3,4,5,6,7,8,9,10,11,12,13,14,15);
            // B fragments (16-bit B 32x16 layout): contiguous 32B row slices
            const __bf16* br0 = &Bbuf[buf][tn2 * 32 + r][k + 16 * half];
            bf16x8 b0l = *(const bf16x8*)br0;
            bf16x8 b0h = *(const bf16x8*)(br0 + 8);
            bf16x16 b0 = __builtin_shufflevector(b0l, b0h,
                         0,1,2,3,4,5,6,7,8,9,10,11,12,13,14,15);
            const __bf16* br1 = &Bbuf[buf][tn2 * 32 + 16 + r][k + 16 * half];
            bf16x8 b1l = *(const bf16x8*)br1;
            bf16x8 b1h = *(const bf16x8*)(br1 + 8);
            bf16x16 b1 = __builtin_shufflevector(b1l, b1h,
                         0,1,2,3,4,5,6,7,8,9,10,11,12,13,14,15);
            acc0 = __builtin_amdgcn_wmma_f32_16x16x32_bf16(false, a, false, b0,
                                                           (short)0, acc0, false, false);
            acc1 = __builtin_amdgcn_wmma_f32_16x16x32_bf16(false, a, false, b1,
                                                           (short)0, acc1, false, false);
        }
        __syncthreads();        // all reads of buf done before it is refilled
        buf ^= 1;
    }

    // Epilogue: D VGPR e -> row m = e + 8*half; fuse cost assembly.
    const int col0 = bn0 + tn2 * 32 + r;
    const int col1 = col0 + 16;
    const float hb0 = 0.5f * nb[col0];
    const float hb1 = 0.5f * nb[col1];
#pragma unroll
    for (int e = 0; e < 8; ++e) {
        const int m = bm0 + tm * 16 + e + 8 * half;
        const float ha = 0.5f * na[m];
        Cout[(size_t)m * NPTS + col0] = ha + hb0 - acc0[e];
        Cout[(size_t)m * NPTS + col1] = ha + hb1 - acc1[e];
    }
}

// ---------------------------------------------------------------------------
// Kernel 3: batched damped softmin (4 independent configs per launch).
//   val    = damp * (-eps * logsumexp_j(base + pot[j]/eps - C[i][j]/eps))
//   out[i] = avg ? 0.5*(avg[i] + val) : val       (fused Sinkhorn averaging)
// transposed=1 reads C^T (serves C_yx = C_xy^T). grid = (512, ncfg).
// ---------------------------------------------------------------------------
struct SmCfg {
    const float* C;
    const float* pot;   // nullable
    const float* avg;   // nullable
    float* out;
    int transposed;
};
struct SmBatch { SmCfg cfg[4]; };

__global__ void softmin4_kernel(SmBatch batch, float base_log, float eps, float dampf) {
    const SmCfg c = batch.cfg[blockIdx.y];
    const int i = blockIdx.x;
    const int t = threadIdx.x;
    const float inv_eps = 1.0f / eps;

    float m = -INFINITY, s = 0.0f;
    for (int j = t; j < NPTS; j += 256) {
        const int idx = c.transposed ? (j * NPTS + i) : (i * NPTS + j);
        float v = base_log - c.C[idx] * inv_eps;
        if (c.pot) v += c.pot[j] * inv_eps;
        if (v > m) { s = s * __expf(m - v) + 1.0f; m = v; }
        else       { s += __expf(v - m); }
    }
    __shared__ float sm[256];
    __shared__ float ss[256];
    sm[t] = m; ss[t] = s;
    __syncthreads();
    for (int w = 128; w > 0; w >>= 1) {
        if (t < w) {
            float m1 = sm[t], s1 = ss[t], m2 = sm[t + w], s2 = ss[t + w];
            float M = fmaxf(m1, m2);
            ss[t] = s1 * __expf(m1 - M) + s2 * __expf(m2 - M);
            sm[t] = M;
        }
        __syncthreads();
    }
    if (t == 0) {
        float val = dampf * (-eps * (sm[0] + __logf(ss[0])));
        c.out[i] = c.avg ? 0.5f * (c.avg[i] + val) : val;
    }
}

// ---------------------------------------------------------------------------
// Kernel 4: unbalanced debiased Sinkhorn cost reduction; zero rest of d_out.
// ---------------------------------------------------------------------------
__global__ void loss_kernel(const float* __restrict__ f_aa, const float* __restrict__ f_ba,
                            const float* __restrict__ g_bb, const float* __restrict__ g_ab,
                            float w, float* __restrict__ out, int out_size) {
    const int t = threadIdx.x;
    const float inv_rho = 1.0f / RHO_C;
    const float inv_n = 1.0f / (float)NPTS;
    float acc = 0.0f;
    for (int i = t; i < NPTS; i += 256) {
        acc += inv_n * (__expf(-f_aa[i] * inv_rho) - __expf(-f_ba[i] * inv_rho));
        acc += inv_n * (__expf(-g_bb[i] * inv_rho) - __expf(-g_ab[i] * inv_rho));
    }
    __shared__ float red[256];
    red[t] = acc;
    __syncthreads();
    for (int s = 128; s > 0; s >>= 1) {
        if (t < s) red[t] += red[t + s];
        __syncthreads();
    }
    if (t == 0) out[0] = w * red[0];
    for (int i = t + 1; i < out_size; i += 256) out[i] = 0.0f;
}

// ---------------------------------------------------------------------------
// Host orchestration: 14 launches total, all graph-capture safe.
// ---------------------------------------------------------------------------
extern "C" void kernel_launch(void* const* d_in, const int* in_sizes, int n_in,
                              void* d_out, int out_size, void* d_ws, size_t ws_size,
                              hipStream_t stream) {
    (void)in_sizes; (void)n_in; (void)ws_size;
    const float* x = (const float*)d_in[1];   // normed_density
    const float* y = (const float*)d_in[2];   // gt
    float* out = (float*)d_out;

    char* ws = (char*)d_ws;
    size_t off = 0;
    auto take = [&](size_t bytes) -> void* {
        off = (off + 255) & ~(size_t)255;
        void* p = ws + off;
        off += bytes;
        return p;
    };
    __bf16* xb = (__bf16*)take((size_t)NPTS * KDIM * sizeof(__bf16));
    __bf16* yb = (__bf16*)take((size_t)NPTS * KDIM * sizeof(__bf16));
    float* Cxy = (float*)take((size_t)NPTS * NPTS * sizeof(float));
    float* Cxx = (float*)take((size_t)NPTS * NPTS * sizeof(float));
    float* Cyy = (float*)take((size_t)NPTS * NPTS * sizeof(float));
    float* nx  = (float*)take(NPTS * sizeof(float));
    float* ny  = (float*)take(NPTS * sizeof(float));
    // double-buffered potentials: [set][0]=f_ba [1]=g_ab [2]=f_aa [3]=g_bb
    float* P[2][4];
    for (int sIdx = 0; sIdx < 2; ++sIdx)
        for (int k = 0; k < 4; ++k) P[sIdx][k] = (float*)take(NPTS * sizeof(float));

    // epsilon schedule: [1] + exp(arange(0, 2*ln .01, 2*ln .5)) + [1e-4]
    float eps_list[16];
    int n_eps = 0;
    eps_list[n_eps++] = 1.0f;
    {
        const double step = 2.0 * log(0.5);
        const double stop = 2.0 * log(0.01);
        for (int k = 0;; ++k) {
            double tt = (double)k * step;
            if (!(tt > stop)) break;
            eps_list[n_eps++] = (float)exp(tt);
        }
    }
    eps_list[n_eps++] = 1e-4f;

    const float base_log = -logf((float)NPTS);    // a_log == b_log (N == M)
    auto damp = [](float eps) { return 1.0f / (1.0f + eps / RHO_C); };

    // stage 1: bf16 convert + squared norms
    convert_norm_kernel<<<2 * NPTS, 256, 0, stream>>>(x, y, xb, yb, nx, ny);

    // stage 2: three cost-matrix GEMMs (LDS-blocked WMMA bf16)
    gemm_cost_kernel<<<dim3(64, 3, 1), 256, 0, stream>>>(xb, yb, nx, ny, Cxy, Cxx, Cyy);

    // stage 3: init at largest blur -> set 0
    int cur = 0;
    {
        const float e0 = eps_list[0], d0 = damp(e0);
        SmBatch b;
        b.cfg[0] = SmCfg{Cxy, nullptr, nullptr, P[cur][0], 0};  // f_ba
        b.cfg[1] = SmCfg{Cxy, nullptr, nullptr, P[cur][1], 1};  // g_ab (C_yx)
        b.cfg[2] = SmCfg{Cxx, nullptr, nullptr, P[cur][2], 0};  // f_aa
        b.cfg[3] = SmCfg{Cyy, nullptr, nullptr, P[cur][3], 0};  // g_bb
        softmin4_kernel<<<dim3(NPTS, 4), 256, 0, stream>>>(b, base_log, e0, d0);
    }

    // stage 4: symmetric updates with epsilon annealing (fused averaging,
    // double-buffered potentials; all four updates read only the old set)
    for (int it = 0; it < n_eps; ++it) {
        const float eps = eps_list[it], d = damp(eps);
        float** oldp = P[cur];
        float** newp = P[cur ^ 1];
        SmBatch b;
        b.cfg[0] = SmCfg{Cxy, oldp[1], oldp[0], newp[0], 0};  // f_ba' <- g_ab
        b.cfg[1] = SmCfg{Cxy, oldp[0], oldp[1], newp[1], 1};  // g_ab' <- f_ba
        b.cfg[2] = SmCfg{Cxx, oldp[2], oldp[2], newp[2], 0};  // f_aa' <- f_aa
        b.cfg[3] = SmCfg{Cyy, oldp[3], oldp[3], newp[3], 0};  // g_bb' <- g_bb
        softmin4_kernel<<<dim3(NPTS, 4), 256, 0, stream>>>(b, base_log, eps, d);
        cur ^= 1;
    }

    // stage 5: final extrapolation at blur^p (no averaging) + loss
    {
        const float eps = eps_list[n_eps - 1], d = damp(eps);
        float** oldp = P[cur];
        float** newp = P[cur ^ 1];
        SmBatch b;
        b.cfg[0] = SmCfg{Cxy, oldp[1], nullptr, newp[0], 0};
        b.cfg[1] = SmCfg{Cxy, oldp[0], nullptr, newp[1], 1};
        b.cfg[2] = SmCfg{Cxx, oldp[2], nullptr, newp[2], 0};
        b.cfg[3] = SmCfg{Cyy, oldp[3], nullptr, newp[3], 0};
        softmin4_kernel<<<dim3(NPTS, 4), 256, 0, stream>>>(b, base_log, eps, d);
        const float w = RHO_C + eps * 0.5f;
        loss_kernel<<<1, 256, 0, stream>>>(newp[2], newp[0], newp[3], newp[1],
                                           w, out, out_size);
    }
}
